// SelfAttention3D_23295902614249
// MI455X (gfx1250) — matrix operand order repaired and checked by reading, not compiled, not run on hardware
//
#include <hip/hip_runtime.h>

// CDNA5 / gfx1250 self-attention: bf16 WMMA GEMMs + fused flash-attention
// with double-buffered async global->LDS staging of K/V tiles.

typedef __attribute__((ext_vector_type(16))) __bf16 v16bf;
typedef __attribute__((ext_vector_type(8)))  float  v8f;

union ABreg { v16bf v; uint4 u[2]; };

// Native conversion: clang emits v_cvt_*bf16_f32 (RTNE) for gfx1250.
__device__ __forceinline__ unsigned short f2bfbits(float f) {
  __bf16 b = (__bf16)f;
  unsigned short h;
  __builtin_memcpy(&h, &b, 2);
  return h;
}

// LDS byte offset of a __shared__ object: ISA maps flat->LDS via addr[31:0].
__device__ __forceinline__ unsigned lds_off(const void* p) {
  return (unsigned)(unsigned long long)p;
}

// Async DMA one 16-byte chunk global->LDS (tracked by ASYNCcnt).
__device__ __forceinline__ void async_ld_b128(unsigned lds, const void* gptr) {
  asm volatile("global_load_async_to_lds_b128 %0, %1, off"
               :: "v"(lds), "v"((unsigned long long)gptr) : "memory");
}

// ---------------------------------------------------------------------------
// Pre-pass A: X fp32 -> bf16, row-major [B*N, 128]. 4 elements per thread.
// ---------------------------------------------------------------------------
__global__ __launch_bounds__(256) void sa3d_cvt_x_kernel(
    const float* __restrict__ X, unsigned short* __restrict__ Xb) {
  const int i = blockIdx.x * 256 + threadIdx.x;  // quad index
  float4 v = ((const float4*)X)[i];
  uint2 o;
  o.x = (unsigned)f2bfbits(v.x) | ((unsigned)f2bfbits(v.y) << 16);
  o.y = (unsigned)f2bfbits(v.z) | ((unsigned)f2bfbits(v.w) << 16);
  ((uint2*)Xb)[i] = o;
}

// ---------------------------------------------------------------------------
// Pre-pass B: W [128,128] fp32 -> Wt [F][C] bf16 (transposed, B-operand
// friendly: per output column, K-dim is contiguous). blockIdx.y picks Q/K/V.
// ---------------------------------------------------------------------------
__global__ __launch_bounds__(256) void sa3d_cvt_w_kernel(
    const float* __restrict__ Wq, const float* __restrict__ Wk,
    const float* __restrict__ Wv, unsigned short* __restrict__ Wt) {
  const float* W = (blockIdx.y == 0) ? Wq : (blockIdx.y == 1) ? Wk : Wv;
  unsigned short* out = Wt + (size_t)blockIdx.y * 128 * 128;
  const int idx = blockIdx.x * 256 + threadIdx.x;  // [0, 16384)
  const int f = idx >> 7;    // output column of GEMM
  const int k = idx & 127;   // contiguous K within row
  out[idx] = f2bfbits(W[(size_t)k * 128 + f]);
}

// ---------------------------------------------------------------------------
// Kernel 1: projection  out = Xb @ W + bias  (all-bf16 operands, f32 acc)
// transpose_out == 0 : out row-major [B*N, 128] bf16   (Q, K)
// transpose_out == 1 : out [B][128][4096] bf16         (V transposed)
// Block = 128 threads (4 waves); wave computes a 16x128 tile, 32 WMMAs.
// ---------------------------------------------------------------------------
__global__ __launch_bounds__(128) void sa3d_proj_kernel(
    const unsigned short* __restrict__ Xb,
    const unsigned short* __restrict__ Wt,
    const float* __restrict__ bias, unsigned short* __restrict__ out,
    int transpose_out) {
  const int lane = threadIdx.x & 31;
  const int wave = threadIdx.x >> 5;
  const int half = lane >> 4;
  const int l16  = lane & 15;
  const int tb   = blockIdx.x * 64 + wave * 16;  // first token (flat over B*N)

  // A operands for all 4 K-chunks: 16x32 bf16 each, pure b128 loads.
  v16bf aq[4];
  {
    const unsigned short* xrow = Xb + (size_t)(tb + l16) * 128;
#pragma unroll
    for (int kc = 0; kc < 4; ++kc) {
      ABreg t;
      t.u[0] = *(const uint4*)(xrow + kc * 32 + half * 8);
      t.u[1] = *(const uint4*)(xrow + kc * 32 + 16 + half * 8);
      aq[kc] = t.v;
    }
  }

  v8f acc[8];
#pragma unroll
  for (int ct = 0; ct < 8; ++ct)
#pragma unroll
    for (int j = 0; j < 8; ++j) acc[ct][j] = 0.0f;

#pragma unroll
  for (int ct = 0; ct < 8; ++ct) {
    const unsigned short* wcol = Wt + (size_t)(ct * 16 + l16) * 128;
#pragma unroll
    for (int kc = 0; kc < 4; ++kc) {
      ABreg bb;  // B operand: K = kc*32 + half*8 + e (e<8), kc*32 + 16 + half*8 + (e-8) (e>=8)
      bb.u[0] = *(const uint4*)(wcol + kc * 32 + half * 8);
      bb.u[1] = *(const uint4*)(wcol + kc * 32 + 16 + half * 8);
      acc[ct] = __builtin_amdgcn_wmma_f32_16x16x32_bf16(
          false, aq[kc], false, bb.v, (short)0, acc[ct], false, false);
    }
  }

  if (!transpose_out) {
    // C layout: VGPR j holds row (j + 8*half), column l16 of the 16x16 tile
#pragma unroll
    for (int ct = 0; ct < 8; ++ct) {
      const int col = ct * 16 + l16;
      const float bv = bias[col];
#pragma unroll
      for (int j = 0; j < 8; ++j) {
        int tok = tb + j + half * 8;
        out[(size_t)tok * 128 + col] = f2bfbits(acc[ct][j] + bv);
      }
    }
  } else {
    const int bidx = tb >> 12;                 // batch (N = 4096)
    const int n0   = (tb & 4095) + half * 8;   // token-in-batch base
#pragma unroll
    for (int ct = 0; ct < 8; ++ct) {
      const int f  = ct * 16 + l16;            // feature index
      const float bv = bias[f];
      unsigned pk[4];
#pragma unroll
      for (int p = 0; p < 4; ++p) {
        unsigned lo = f2bfbits(acc[ct][2 * p]     + bv);
        unsigned hi = f2bfbits(acc[ct][2 * p + 1] + bv);
        pk[p] = lo | (hi << 16);
      }
      // 8 consecutive tokens for fixed feature -> one b128 store
      *(uint4*)(out + ((size_t)bidx * 128 + f) * 4096 + n0) =
          make_uint4(pk[0], pk[1], pk[2], pk[3]);
    }
  }
}

// ---------------------------------------------------------------------------
// Kernel 2: fused flash attention with async K/V staging.
// Q,K: [B*N,128] bf16 row-major. Vt: [B][128][4096] bf16. Out: [B*N,128] fp32.
// Block = 128 threads (4 waves); each wave owns 16 query rows. The block
// cooperatively DMAs each 32-key K/V tile into LDS (double-buffered,
// global_load_async_to_lds_b128 + s_wait_asynccnt) so the 4 waves share it.
// ---------------------------------------------------------------------------
__global__ __launch_bounds__(128) void sa3d_attn_kernel(
    const unsigned short* __restrict__ Q, const unsigned short* __restrict__ K,
    const unsigned short* __restrict__ Vt, float* __restrict__ Out) {
  __shared__ __align__(16) unsigned short Kl[2][32][128];   // 2 x 8 KB
  __shared__ __align__(16) unsigned short Vl[2][128][32];   // 2 x 8 KB
  __shared__ __align__(16) unsigned short Plds[4][16][32];  // 4 KB

  const int tid  = threadIdx.x;
  const int lane = tid & 31;
  const int wave = tid >> 5;
  const int half = lane >> 4;
  const int l16  = lane & 15;
  const int tb   = blockIdx.x * 64 + wave * 16;  // first query token (flat)
  const int bidx = tb >> 12;
  const float scale = 0.08838834764831845f;      // 1/sqrt(128)

  const unsigned short* Kbase = K  + (size_t)bidx * 4096 * 128;
  const unsigned short* Vbase = Vt + (size_t)bidx * 128 * 4096;

  // Cooperative async stage of tile kt_ into buffer buf: 8 b128 DMAs/thread.
  auto stage = [&](int kt_, int buf) {
    // K tile: 32 rows x 256 B. thread -> row (tid&31), 64-B quarter (tid>>5).
    const int r = tid & 31, q = tid >> 5;
    const unsigned short* gk = Kbase + (size_t)(kt_ * 32 + r) * 128 + q * 32;
    unsigned lk = lds_off(&Kl[buf][r][q * 32]);
#pragma unroll
    for (int i = 0; i < 4; ++i) async_ld_b128(lk + i * 16, gk + i * 8);
    // V tile: 128 rows (features) x 64 B. thread -> one feature row.
    const unsigned short* gv = Vbase + (size_t)tid * 4096 + kt_ * 32;
    unsigned lv = lds_off(&Vl[buf][tid][0]);
#pragma unroll
    for (int i = 0; i < 4; ++i) async_ld_b128(lv + i * 16, gv + i * 8);
  };

  // Q tile as 4 A-operands (16x32 each), resident for the whole loop.
  v16bf aq[4];
  {
    const unsigned short* qrow = Q + (size_t)(tb + l16) * 128;
#pragma unroll
    for (int kc = 0; kc < 4; ++kc) {
      ABreg t;
      t.u[0] = *(const uint4*)(qrow + kc * 32 + half * 8);
      t.u[1] = *(const uint4*)(qrow + kc * 32 + 16 + half * 8);
      aq[kc] = t.v;
    }
  }

  float mrow[8], lrow[8];
  v8f accO[8];
#pragma unroll
  for (int j = 0; j < 8; ++j) { mrow[j] = -1e30f; lrow[j] = 0.0f; }
#pragma unroll
  for (int ct = 0; ct < 8; ++ct)
#pragma unroll
    for (int j = 0; j < 8; ++j) accO[ct][j] = 0.0f;

  stage(0, 0);  // prologue DMA

  for (int kt = 0; kt < 128; ++kt) {
    const int buf = kt & 1;
    if (kt + 1 < 128) {
      stage(kt + 1, buf ^ 1);  // overlap next tile's DMA with this compute
      // 16 async ops outstanding; drain the older 8 (in-order completion).
      asm volatile("s_wait_asynccnt 8" ::: "memory");
    } else {
      asm volatile("s_wait_asynccnt 0" ::: "memory");
    }
    __syncthreads();  // current tile visible to all waves

    // ---- S = Q @ K^T for 32 keys (two 16x16 column tiles), K from LDS ----
    v8f s[2];
#pragma unroll
    for (int t = 0; t < 2; ++t) {
#pragma unroll
      for (int j = 0; j < 8; ++j) s[t][j] = 0.0f;
      const unsigned short* krow = &Kl[buf][t * 16 + l16][0];
#pragma unroll
      for (int kc = 0; kc < 4; ++kc) {
        ABreg bb;
        bb.u[0] = *(const uint4*)(krow + kc * 32 + half * 8);
        bb.u[1] = *(const uint4*)(krow + kc * 32 + 16 + half * 8);
        s[t] = __builtin_amdgcn_wmma_f32_16x16x32_bf16(
            false, aq[kc], false, bb.v, (short)0, s[t], false, false);
      }
    }

    // ---- online softmax over the 32 keys ----
    float rmax[8];
#pragma unroll
    for (int j = 0; j < 8; ++j) {
      s[0][j] *= scale; s[1][j] *= scale;
      rmax[j] = fmaxf(s[0][j], s[1][j]);
    }
#pragma unroll
    for (int off = 8; off >= 1; off >>= 1)
#pragma unroll
      for (int j = 0; j < 8; ++j)
        rmax[j] = fmaxf(rmax[j], __shfl_xor(rmax[j], off, 32));

    float alpha[8], rsum[8];
#pragma unroll
    for (int j = 0; j < 8; ++j) {
      float mnew = fmaxf(mrow[j], rmax[j]);
      alpha[j] = __expf(mrow[j] - mnew);
      mrow[j] = mnew;
      s[0][j] = __expf(s[0][j] - mnew);
      s[1][j] = __expf(s[1][j] - mnew);
      rsum[j] = s[0][j] + s[1][j];
    }
#pragma unroll
    for (int off = 8; off >= 1; off >>= 1)
#pragma unroll
      for (int j = 0; j < 8; ++j)
        rsum[j] += __shfl_xor(rsum[j], off, 32);
#pragma unroll
    for (int j = 0; j < 8; ++j) lrow[j] = lrow[j] * alpha[j] + rsum[j];
#pragma unroll
    for (int ct = 0; ct < 8; ++ct)
#pragma unroll
      for (int j = 0; j < 8; ++j) accO[ct][j] *= alpha[j];

    // ---- transpose P through LDS: C-layout -> A-operand layout ----
#pragma unroll
    for (int t = 0; t < 2; ++t)
#pragma unroll
      for (int j = 0; j < 8; ++j)
        Plds[wave][j + 8 * half][t * 16 + l16] = f2bfbits(s[t][j]);
    __syncthreads();

    ABreg ap;
    ap.u[0] = *(const uint4*)&Plds[wave][l16][half * 8];
    ap.u[1] = *(const uint4*)&Plds[wave][l16][16 + half * 8];

    // ---- O += P @ V, V tile from LDS (K-dim contiguous ds_load_b128) ----
#pragma unroll
    for (int ct = 0; ct < 8; ++ct) {
      const unsigned short* vrow = &Vl[buf][ct * 16 + l16][half * 8];
      ABreg bb;
      bb.u[0] = *(const uint4*)(vrow);
      bb.u[1] = *(const uint4*)(vrow + 16);
      accO[ct] = __builtin_amdgcn_wmma_f32_16x16x32_bf16(
          false, ap.v, false, bb.v, (short)0, accO[ct], false, false);
    }
    __syncthreads();  // all reads of buf done before it is re-staged
  }

  // ---- epilogue: divide by running sum, store fp32 ----
#pragma unroll
  for (int ct = 0; ct < 8; ++ct) {
    const int col = ct * 16 + l16;
#pragma unroll
    for (int j = 0; j < 8; ++j) {
      int tok = tb + j + half * 8;
      Out[(size_t)tok * 128 + col] = accO[ct][j] / lrow[j];
    }
  }
}

// ---------------------------------------------------------------------------
extern "C" void kernel_launch(void* const* d_in, const int* in_sizes, int n_in,
                              void* d_out, int out_size, void* d_ws,
                              size_t ws_size, hipStream_t stream) {
  const float* X  = (const float*)d_in[0];
  const float* Wq = (const float*)d_in[1];
  const float* bq = (const float*)d_in[2];
  const float* Wk = (const float*)d_in[3];
  const float* bk = (const float*)d_in[4];
  const float* Wv = (const float*)d_in[5];
  const float* bv = (const float*)d_in[6];

  const size_t tokf = (size_t)4 * 4096 * 128;  // B*N*F elements
  unsigned short* qs  = (unsigned short*)d_ws;         // 4 MB
  unsigned short* ks  = qs  + tokf;                    // 4 MB
  unsigned short* vts = ks  + tokf;                    // 4 MB
  unsigned short* xb  = vts + tokf;                    // 4 MB
  unsigned short* wt  = xb  + tokf;                    // 96 KB (3 x 128x128)

  // Pre-convert activations and weights to bf16 (weights transposed).
  sa3d_cvt_x_kernel<<<2048, 256, 0, stream>>>(X, xb);        // 2M elems / 4
  sa3d_cvt_w_kernel<<<dim3(64, 3), 256, 0, stream>>>(Wq, Wk, Wv, wt);

  // 16384 tokens / 64 per block = 256 blocks, 128 threads (4 wave32s)
  sa3d_proj_kernel<<<256, 128, 0, stream>>>(xb, wt,             bq, qs, 0);
  sa3d_proj_kernel<<<256, 128, 0, stream>>>(xb, wt + 128 * 128, bk, ks, 0);
  sa3d_proj_kernel<<<256, 128, 0, stream>>>(xb, wt + 2 * 128 * 128, bv, vts, 1);
  sa3d_attn_kernel<<<256, 128, 0, stream>>>(qs, ks, vts, (float*)d_out);
}
